// RNN_72584947302566
// MI455X (gfx1250) — matrix operand order repaired and checked by reading, not compiled, not run on hardware
//
#include <hip/hip_runtime.h>
#include <hip/hip_bf16.h>

// ---------------------------------------------------------------------------
// RNN (2-layer tanh) for MI455X / gfx1250.
// bf16 WMMA (v_wmma_f32_16x16x32_bf16) + TDM (tensor_load_to_lds) A-staging,
// software-pipelined 2 tiles ahead across 4 LDS buffers.
// ---------------------------------------------------------------------------

typedef __attribute__((ext_vector_type(16))) __bf16        v16bf;
typedef __attribute__((ext_vector_type(8)))  float         v8f;
typedef __attribute__((ext_vector_type(4)))  unsigned int  v4u;
typedef __attribute__((ext_vector_type(4)))  int           v4i;
typedef __attribute__((ext_vector_type(8)))  int           v8i;

#define RNN_S 512
#define RNN_B 64
#define RNN_I 256
#define RNN_H 1024
#define RNN_L 2

union FragBF { v16bf v; unsigned int d[8]; v4u q[2]; };

__device__ __forceinline__ unsigned short f2bf(float f) {
  unsigned u = __float_as_uint(f);
  unsigned r = u + 0x7FFFu + ((u >> 16) & 1u);   // round-to-nearest-even
  return (unsigned short)(r >> 16);
}
__device__ __forceinline__ unsigned pack2bf(float lo, float hi) {
  return (unsigned)f2bf(lo) | ((unsigned)f2bf(hi) << 16);
}

// ---------------------------------------------------------------------------
// TDM: DMA one 16x32 bf16 tile (row stride K elements) from global into LDS.
// D# per CDNA5 ISA ch.8: group0 = {count=1, lds_addr, global_addr, type=2},
// group1 = {data_size=2B, tensor_dim0=K, tensor_dim1=Mrows, tile=32x16,
//           dim0_stride=K}. Groups 2/3 zero (2D tensor).
// ---------------------------------------------------------------------------
__device__ __forceinline__ void tdm_load_tile_a(const unsigned short* gptr,
                                                unsigned lds_off, int K,
                                                int Mrows) {
  unsigned long long ga = (unsigned long long)gptr;
  v4u g0;
  g0[0] = 1u;                                     // count=1, no gather
  g0[1] = lds_off;                                // LDS byte address
  g0[2] = (unsigned)ga;                           // global addr [31:0]
  g0[3] = (unsigned)(ga >> 32) | (2u << 30);      // addr [56:32] | type=2
  v8i g1;
  g1[0] = (int)(1u << 16);                        // data_size = 2 bytes
  g1[1] = (int)(((unsigned)K & 0xFFFFu) << 16);   // tensor_dim0[15:0]
  g1[2] = (int)(((unsigned)K >> 16) | (((unsigned)Mrows & 0xFFFFu) << 16));
  g1[3] = (int)(((unsigned)Mrows >> 16) | (32u << 16));   // tile_dim0 = 32
  g1[4] = 16;                                     // tile_dim1=16, tile_dim2=0
  g1[5] = K;                                      // tensor_dim0_stride
  g1[6] = 0;
  g1[7] = 0;
  v4i z4 = {0, 0, 0, 0};
#if __has_include(<hip/amd_detail/amd_gfx1250_TDM.h>)
  v8i z8 = {0, 0, 0, 0, 0, 0, 0, 0};
  __builtin_amdgcn_tensor_load_to_lds(g0, g1, z4, z4, z8, 0);  // clang-23 form
#else
  __builtin_amdgcn_tensor_load_to_lds(g0, g1, z4, z4, 0);      // ROCm 7.2 form
#endif
}

// ---------------------------------------------------------------------------
// fp32 -> bf16 elementwise (grid-stride)
// ---------------------------------------------------------------------------
__global__ void k_cvt_bf16(const float* __restrict__ src,
                           unsigned short* __restrict__ dst, size_t n) {
  size_t i = (size_t)blockIdx.x * blockDim.x + threadIdx.x;
  size_t stride = (size_t)gridDim.x * blockDim.x;
  for (; i < n; i += stride) dst[i] = f2bf(src[i]);
}

// ---------------------------------------------------------------------------
// Swizzle fp32 W[N,K] (row-major, used as B^T) into WMMA B-fragment order,
// LANE-MAJOR: Bsw[(ntile*KT + kt)*256 + lane*8 + j], so each lane fetches its
// 8 fragment dwords as two contiguous b128 loads (wave reads 1KB run).
// B-matrix 32x16 bf16 layout: lane = 16*half + n, dword j <-> K = 2(j&3) + 8*half + 16*(j>>2).
// ---------------------------------------------------------------------------
__global__ void k_swizzleB(const float* __restrict__ W,
                           unsigned* __restrict__ Bsw, int N, int K) {
  int idx = blockIdx.x * blockDim.x + threadIdx.x;
  int total = (N >> 4) * (K >> 5) * 256;
  if (idx >= total) return;
  int j    = idx & 7;
  int lane = (idx >> 3) & 31;
  int tile = idx >> 8;
  int KT = K >> 5;
  int ntile = tile / KT, kt = tile % KT;
  int half = lane >> 4;
  int n = ntile * 16 + (lane & 15);
  int k = kt * 32 + 2 * (j & 3) + 8 * half + 16 * (j >> 2);
  Bsw[idx] = pack2bf(W[(size_t)n * K + k], W[(size_t)n * K + k + 1]);
}

// ---------------------------------------------------------------------------
// C[M,N] = A[M,K](bf16 row-major) @ Bsw^T + bias[N]   (fp32 out)
// block = 256 thr (8 waves) owning (mtile, 8 ntiles).
// A 16x32 tiles DMA'd by the TDM into 4 LDS buffers, issued 2 k-steps ahead
// by wave 0 (steady-state wait: tensorcnt<=2). Issue-distance 2 means the DMA
// target buffer was last read two iterations ago, so the previous iteration's
// barrier already proves those reads completed -> ONE barrier per iteration,
// no conditionals in the steady-state body (last two iterations peeled).
// ---------------------------------------------------------------------------
__global__ __launch_bounds__(256) void k_gemm_bf16(
    const unsigned short* __restrict__ A, const unsigned* __restrict__ Bsw,
    const float* __restrict__ bias, float* __restrict__ C,
    int M, int N, int K) {
  __shared__ __align__(16) unsigned short Atile[4][16 * 32];
  const int KT = K >> 5;                        // 8 or 32 (>= 4)
  const int ngrp_per = N >> 7;
  const int mtile = blockIdx.x / ngrp_per;
  const int ngrp  = blockIdx.x % ngrp_per;
  const int wave = threadIdx.x >> 5;
  const int lane = threadIdx.x & 31;
  const int ntile = ngrp * 8 + wave;
  const int half = lane >> 4;
  const int ln15 = lane & 15;
  const int m0 = mtile * 16;
  const int n0 = ntile * 16;
  const unsigned short* Ablk = A + (size_t)m0 * K;   // block's 16-row panel
  const unsigned ldsbase = (unsigned)(unsigned long long)(const void*)&Atile[0][0];

  v8f acc = {};
  auto body = [&](int kt) {
    FragBF a, b;
    const unsigned short* at = &Atile[kt & 3][ln15 * 32 + 8 * half];
    a.q[0] = *(const v4u*)at;                    // ds_load_b128 x2
    a.q[1] = *(const v4u*)(at + 16);
    const v4u* bp = (const v4u*)(Bsw + (size_t)(ntile * KT + kt) * 256 + lane * 8);
    b.q[0] = bp[0];                              // global_load_b128 x2
    b.q[1] = bp[1];
    acc = __builtin_amdgcn_wmma_f32_16x16x32_bf16(false, a.v, false, b.v,
                                                  (short)0, acc, false, false);
  };

  if (wave == 0) {   // prologue: DMA tiles 0 and 1
    tdm_load_tile_a(Ablk,      ldsbase,        K, M);
    tdm_load_tile_a(Ablk + 32, ldsbase + 1024, K, M);
  }
  for (int kt = 0; kt < KT - 2; ++kt) {          // steady state
    if (wave == 0) {
      tdm_load_tile_a(Ablk + (size_t)(kt + 2) * 32,
                      ldsbase + ((kt + 2) & 3) * 1024, K, M);
      __builtin_amdgcn_s_wait_tensorcnt(2);      // tile kt complete
    }
    __syncthreads();
    __builtin_prefetch(&Bsw[(size_t)(ntile * KT + kt + 1) * 256 + lane * 8], 0, 1);
    body(kt);
  }
  if (wave == 0) __builtin_amdgcn_s_wait_tensorcnt(1);
  __syncthreads();
  body(KT - 2);
  if (wave == 0) __builtin_amdgcn_s_wait_tensorcnt(0);
  __syncthreads();
  body(KT - 1);

  const float bn = bias[n0 + ln15];
#pragma unroll
  for (int r = 0; r < 8; ++r) {            // C layout: VGPR r -> M = r + 8*half
    int m = m0 + r + 8 * half;
    C[(size_t)m * N + n0 + ln15] = acc[r] + bn;
  }
}

// ---------------------------------------------------------------------------
// init scan state: h0 = 0 (both ping-pong bf16 buffers), barrier counters = 0
// ---------------------------------------------------------------------------
__global__ void k_scan_init(unsigned short* h16a, unsigned short* h16b, int n,
                            int* sync) {
  int i = blockIdx.x * blockDim.x + threadIdx.x;
  if (i < n) { h16a[i] = 0; h16b[i] = 0; }
  if (i == 0) { sync[0] = 0; sync[1] = 0; }
}

// ---------------------------------------------------------------------------
// Persistent recurrent scan: for t: h = tanh(pre[t] + h @ Whh^T + b_hh)
// Grid = 32 blocks x 8 waves = 256 waves = 4 mtiles x 64 ntiles.
// Ping-pong bf16 h buffers + software grid barrier between steps.
// ---------------------------------------------------------------------------
__global__ __launch_bounds__(256) void k_rnn_scan(
    const unsigned* __restrict__ Whh_sw, const float* __restrict__ pre,
    const float* __restrict__ bhh, float* __restrict__ act32,
    unsigned short* __restrict__ h16a, unsigned short* __restrict__ h16b,
    float* __restrict__ hid_out, int* __restrict__ sync) {
  const int wid  = blockIdx.x * 8 + (threadIdx.x >> 5);
  const int lane = threadIdx.x & 31;
  const int mtile = wid >> 6;
  const int ntile = wid & 63;
  const int half = lane >> 4;
  const int ln15 = lane & 15;
  const int m0 = mtile * 16, n0 = ntile * 16;
  const float bn = bhh[n0 + ln15];
  int* cnt = sync;
  int* gen = sync + 1;

  for (int t = 0; t < RNN_S; ++t) {
    const unsigned short* hsrc = (t & 1) ? h16b : h16a;
    unsigned short* hdst = (t & 1) ? h16a : h16b;
    v8f acc = {};
#pragma unroll 2
    for (int kt = 0; kt < RNN_H / 32; ++kt) {
      FragBF a, b;
      // branch-free next-tile prefetch hint (wraps to 0 on last iter)
      __builtin_prefetch(
          &Whh_sw[(size_t)(ntile * 32 + ((kt + 1) & 31)) * 256 + lane * 8], 0, 1);
      // A frag from L2-resident h (row-major): two global b128 per lane
      const unsigned short* ap =
          hsrc + (size_t)(m0 + ln15) * RNN_H + kt * 32 + 8 * half;
      a.q[0] = *(const v4u*)ap;
      a.q[1] = *(const v4u*)(ap + 16);
      // B frag: two global b128 per lane
      const v4u* bp = (const v4u*)(Whh_sw + (size_t)(ntile * 32 + kt) * 256 + lane * 8);
      b.q[0] = bp[0];
      b.q[1] = bp[1];
      acc = __builtin_amdgcn_wmma_f32_16x16x32_bf16(false, a.v, false, b.v,
                                                    (short)0, acc, false, false);
    }
    const size_t base = (size_t)t * RNN_B * RNN_H;
#pragma unroll
    for (int r = 0; r < 8; ++r) {
      int m = m0 + r + 8 * half;
      size_t idx = base + (size_t)m * RNN_H + n0 + ln15;
      float v = tanhf(acc[r] + pre[idx] + bn);
      act32[idx] = v;
      hdst[(size_t)m * RNN_H + n0 + ln15] = f2bf(v);
      if (t == RNN_S - 1) hid_out[(size_t)m * RNN_H + n0 + ln15] = v;
    }
    // ---- software grid barrier ----
    __threadfence();
    __syncthreads();
    if (threadIdx.x == 0) {
      int g = __hip_atomic_load(gen, __ATOMIC_RELAXED, __HIP_MEMORY_SCOPE_AGENT);
      int arrived =
          __hip_atomic_fetch_add(cnt, 1, __ATOMIC_ACQ_REL, __HIP_MEMORY_SCOPE_AGENT);
      if (arrived == (int)gridDim.x - 1) {
        __hip_atomic_store(cnt, 0, __ATOMIC_RELAXED, __HIP_MEMORY_SCOPE_AGENT);
        __hip_atomic_fetch_add(gen, 1, __ATOMIC_RELEASE, __HIP_MEMORY_SCOPE_AGENT);
      } else {
        while (__hip_atomic_load(gen, __ATOMIC_ACQUIRE,
                                 __HIP_MEMORY_SCOPE_AGENT) == g)
          __builtin_amdgcn_s_sleep(1);
      }
    }
    __syncthreads();
  }
}

// ---------------------------------------------------------------------------
extern "C" void kernel_launch(void* const* d_in, const int* in_sizes, int n_in,
                              void* d_out, int out_size, void* d_ws,
                              size_t ws_size, hipStream_t stream) {
  (void)in_sizes; (void)n_in; (void)out_size; (void)ws_size;
  const int S = RNN_S, B = RNN_B, I = RNN_I, H = RNN_H;

  const float* x     = (const float*)d_in[0];  // [S,B,I]
  const float* W_in  = (const float*)d_in[1];  // [H,I]
  const float* b_in  = (const float*)d_in[2];  // [H]
  const float* W_ih  = (const float*)d_in[3];  // [L,H,H]
  const float* W_hh  = (const float*)d_in[4];  // [L,H,H]
  const float* b_ih  = (const float*)d_in[5];  // [L,H]
  const float* b_hh  = (const float*)d_in[6];  // [L,H]
  const float* W_out = (const float*)d_in[7];  // [I,H]
  const float* b_out = (const float*)d_in[8];  // [I]
  float* out = (float*)d_out;                  // [S,B,I] ++ [L,B,H]

  char* ws = (char*)d_ws;
  size_t o = 0;
  auto take = [&](size_t bytes) { size_t r = o; o += (bytes + 255) & ~(size_t)255; return r; };
  float*          act32  = (float*)(ws + take((size_t)S * B * H * 4));
  float*          pre32  = (float*)(ws + take((size_t)S * B * H * 4));
  unsigned short* act16  = (unsigned short*)(ws + take((size_t)S * B * H * 2));
  unsigned*       winsw  = (unsigned*)(ws + take((size_t)H * I * 2));
  unsigned*       wihsw  = (unsigned*)(ws + take((size_t)H * H * 2));
  unsigned*       whhsw  = (unsigned*)(ws + take((size_t)H * H * 2));
  unsigned*       woutsw = (unsigned*)(ws + take((size_t)I * H * 2));
  unsigned short* h16a   = (unsigned short*)(ws + take((size_t)B * H * 2));
  unsigned short* h16b   = (unsigned short*)(ws + take((size_t)B * H * 2));
  int*            syncp  = (int*)(ws + take(256));
  unsigned short* x16 = act16;  // x16 (16.8MB) lives in act16 region (67MB)

  const int M = S * B;  // 32768 rows for all big GEMMs

  k_cvt_bf16<<<8192, 256, 0, stream>>>(x, x16, (size_t)S * B * I);
  k_swizzleB<<<((H / 16) * (I / 32) * 256) / 256, 256, 0, stream>>>(W_in, winsw, H, I);
  k_gemm_bf16<<<(M / 16) * (H / 128), 256, 0, stream>>>(x16, winsw, b_in, act32, M, H, I);

  for (int l = 0; l < RNN_L; ++l) {
    k_swizzleB<<<((H / 16) * (H / 32) * 256) / 256, 256, 0, stream>>>(
        W_ih + (size_t)l * H * H, wihsw, H, H);
    k_swizzleB<<<((H / 16) * (H / 32) * 256) / 256, 256, 0, stream>>>(
        W_hh + (size_t)l * H * H, whhsw, H, H);
    k_cvt_bf16<<<16384, 256, 0, stream>>>(act32, act16, (size_t)S * B * H);
    k_gemm_bf16<<<(M / 16) * (H / 128), 256, 0, stream>>>(
        act16, wihsw, b_ih + (size_t)l * H, pre32, M, H, H);
    k_scan_init<<<(B * H + 255) / 256, 256, 0, stream>>>(h16a, h16b, B * H, syncp);
    k_rnn_scan<<<32, 256, 0, stream>>>(whhsw, pre32, b_hh + (size_t)l * H, act32,
                                       h16a, h16b,
                                       out + (size_t)S * B * I + (size_t)l * B * H,
                                       syncp);
  }

  k_cvt_bf16<<<16384, 256, 0, stream>>>(act32, act16, (size_t)S * B * H);
  k_swizzleB<<<((I / 16) * (H / 32) * 256) / 256, 256, 0, stream>>>(W_out, woutsw, I, H);
  k_gemm_bf16<<<(M / 16) * (I / 128), 256, 0, stream>>>(act16, woutsw, b_out, out, M, I, H);
}
